// MultiHeadAttentionBlock_24764781429184
// MI455X (gfx1250) — matrix operand order repaired and checked by reading, not compiled, not run on hardware
//
#include <hip/hip_runtime.h>
#include <hip/hip_bf16.h>

// MultiHeadAttentionBlock for MI455X (gfx1250), bf16 WMMA path.
// Workspace layout (requires ~42 MB):
//   [0,8MB)    xn  bf16 [B=8][N=1024][C=512]
//   [8,16MB)   q   bf16 [B][h=8][N][d=64]   (pre-scaled by 1/sqrt(d))
//   [16,24MB)  k   bf16 [B][h][N][d]
//   [24,32MB)  v   bf16 [B][h][d][N]        (transposed for P*V)
//   [32,40MB)  ctx bf16 [B][N][C]
//   [40MB,..)  bf16 copies of wq,wk,wv,wo (0.5MB each)

typedef __bf16 bf16;
typedef __attribute__((ext_vector_type(16))) __bf16 v16bf;
typedef __attribute__((ext_vector_type(8)))  __bf16 v8bf;
typedef __attribute__((ext_vector_type(8)))  float  v8f;

// ---- gfx1250 async global->LDS path (ASYNCcnt), guarded so compile never breaks
#if defined(__has_builtin)
#if __has_builtin(__builtin_amdgcn_global_load_async_to_lds_b128) && \
    __has_builtin(__builtin_amdgcn_s_wait_asynccnt)
#define USE_ASYNC_LDS 1
#endif
#endif
#ifndef USE_ASYNC_LDS
#define USE_ASYNC_LDS 0
#endif

// Parameter type of the async-load builtin (from the compiler diagnostic):
// "int __vector(4) __device__ *"  -> vector_size(16) int, AS(1)/AS(3) pointers.
typedef int b128vec __attribute__((vector_size(16)));
typedef __attribute__((address_space(1))) b128vec* gbl_b128_ptr;
typedef __attribute__((address_space(3))) b128vec* lds_b128_ptr;

static __device__ __forceinline__ void copy16_g2l(const bf16* g, bf16* l) {
#if USE_ASYNC_LDS
  __builtin_amdgcn_global_load_async_to_lds_b128(
      (gbl_b128_ptr)g, (lds_b128_ptr)l, 0, 0);
#else
  *(v8bf*)l = *(const v8bf*)g;
#endif
}

static __device__ __forceinline__ void async_copy_wait() {
#if USE_ASYNC_LDS
  __builtin_amdgcn_s_wait_asynccnt(0);
#endif
}

static __device__ __forceinline__ bf16 to_bf16(float f) {
  unsigned u = __builtin_bit_cast(unsigned, f);
  u += 0x7fffu + ((u >> 16) & 1u);              // round-to-nearest-even
  return __builtin_bit_cast(bf16, (unsigned short)(u >> 16));
}

static __device__ __forceinline__ v8f wmma_bf16(v16bf a, v16bf b, v8f c) {
  return __builtin_amdgcn_wmma_f32_16x16x32_bf16(false, a, false, b, (short)0, c,
                                                 false, false);
}

// A fragment 16x32 (MxK): source row-major, K contiguous, row stride ld elems.
// Lane L: row = L&15; lanes 0-15 hold K {0..7,16..23}, lanes 16-31 K {8..15,24..31}.
static __device__ __forceinline__ v16bf load_frag_a(const bf16* base, int ld) {
  int lane = threadIdx.x & 31;
  int row = lane & 15, hi = lane >> 4;
  const bf16* p = base + row * ld + 8 * hi;
  v8bf l0 = *(const v8bf*)p;
  v8bf l1 = *(const v8bf*)(p + 16);
  v16bf r;
#pragma unroll
  for (int e = 0; e < 8; ++e) { r[e] = l0[e]; r[e + 8] = l1[e]; }
  return r;
}

// B fragment 32x16 (KxN): source is B^T row-major (rows = N), K contiguous.
// Lane L: col N = L&15; lanes 0-15 hold K {0..7,16..23}, lanes 16-31 K {8..15,24..31}.
static __device__ __forceinline__ v16bf load_frag_b(const bf16* base, int ld) {
  int lane = threadIdx.x & 31;
  int n = lane & 15, hi = lane >> 4;
  const bf16* p = base + n * ld + 8 * hi;
  v8bf l0 = *(const v8bf*)p;
  v8bf l1 = *(const v8bf*)(p + 16);
  v16bf r;
#pragma unroll
  for (int e = 0; e < 8; ++e) { r[e] = l0[e]; r[e + 8] = l1[e]; }
  return r;
}

// ---------------- Kernel 0: weights fp32 -> bf16 ----------------
__global__ __launch_bounds__(256) void wcvt_kernel(
    const float* __restrict__ a, const float* __restrict__ b,
    const float* __restrict__ c, const float* __restrict__ d,
    bf16* __restrict__ oa, bf16* __restrict__ ob,
    bf16* __restrict__ oc, bf16* __restrict__ od) {
  int i = blockIdx.x * 256 + threadIdx.x;
  if (i < 512 * 512) {
    oa[i] = to_bf16(a[i]);
    ob[i] = to_bf16(b[i]);
    oc[i] = to_bf16(c[i]);
    od[i] = to_bf16(d[i]);
  }
}

// ---------------- Kernel 1: GroupNorm -> xn bf16 [B][N][C] ----------------
__global__ __launch_bounds__(256) void gn_kernel(const float* __restrict__ x,
                                                 const float* __restrict__ gw,
                                                 const float* __restrict__ gb,
                                                 bf16* __restrict__ xn) {
  const int b = blockIdx.x >> 5;     // 8 batches
  const int g = blockIdx.x & 31;     // 32 groups
  const int c0 = g * 16;
  const float* xp = x + ((size_t)b * 512 + c0) * 1024;  // 16 ch x 1024 contiguous
  float s = 0.f, s2 = 0.f;
  for (int i = threadIdx.x; i < 16384; i += 256) {
    float vv = xp[i]; s += vv; s2 += vv * vv;
  }
#pragma unroll
  for (int mk = 16; mk >= 1; mk >>= 1) {
    s  += __shfl_xor(s, mk, 32);
    s2 += __shfl_xor(s2, mk, 32);
  }
  __shared__ float red[16];
  __shared__ float stats[2];
  const int w = threadIdx.x >> 5, lane = threadIdx.x & 31;
  if (lane == 0) { red[w] = s; red[8 + w] = s2; }
  __syncthreads();
  if (threadIdx.x == 0) {
    float ts = 0.f, t2 = 0.f;
    for (int i = 0; i < 8; ++i) { ts += red[i]; t2 += red[8 + i]; }
    float mu = ts * (1.f / 16384.f);
    float var = t2 * (1.f / 16384.f) - mu * mu;
    stats[0] = mu;
    stats[1] = rsqrtf(var + 1e-5f);
  }
  __syncthreads();
  const float mu = stats[0], rstd = stats[1];
  for (int i = threadIdx.x; i < 16384; i += 256) {
    int cl = i >> 10, n = i & 1023;
    int c = c0 + cl;
    float vv = (xp[i] - mu) * rstd * gw[c] + gb[c];
    xn[((size_t)b * 1024 + n) * 512 + c] = to_bf16(vv);
  }
}

// ---------------- Kernel 2: QKV projection GEMM ----------------
// Y[t][o] = sum_c xn[t][c] * W[o][c] + bias[o];  M=8192, N=512, K=512.
// MODE 0: q (x 1/8), MODE 1: k, MODE 2: v (transposed store).
template <int MODE>
__global__ __launch_bounds__(256) void qkv_gemm_kernel(
    const bf16* __restrict__ xn, const bf16* __restrict__ wt,
    const float* __restrict__ bias, bf16* __restrict__ dst) {
  __shared__ __align__(16) bf16 As[128 * 32];
  __shared__ __align__(16) bf16 Bs[128 * 32];
  const int tid = threadIdx.x;
  const int m0 = (blockIdx.x >> 2) * 128;   // 64 M tiles
  const int n0 = (blockIdx.x & 3) * 128;    // 4  N tiles
  const int w  = tid >> 5;
  const int wm = w & 1;                     // 2 waves along M (64 rows)
  const int wn = w >> 1;                    // 4 waves along N (32 cols)
  v8f acc[4][2];
#pragma unroll
  for (int i = 0; i < 4; ++i)
#pragma unroll
    for (int j = 0; j < 2; ++j)
#pragma unroll
      for (int e = 0; e < 8; ++e) acc[i][j][e] = 0.f;

  for (int kk = 0; kk < 512; kk += 32) {
    __syncthreads();
    for (int ch = tid; ch < 512; ch += 256) {
      int r = ch >> 2, c8 = (ch & 3) * 8;
      copy16_g2l(&xn[(size_t)(m0 + r) * 512 + kk + c8], &As[r * 32 + c8]);
      copy16_g2l(&wt[(size_t)(n0 + r) * 512 + kk + c8], &Bs[r * 32 + c8]);
    }
    async_copy_wait();
    __syncthreads();
    v16bf bf0 = load_frag_b(&Bs[(wn * 32 + 0) * 32], 32);
    v16bf bf1 = load_frag_b(&Bs[(wn * 32 + 16) * 32], 32);
#pragma unroll
    for (int it = 0; it < 4; ++it) {
      v16bf a = load_frag_a(&As[(wm * 64 + it * 16) * 32], 32);
      acc[it][0] = wmma_bf16(a, bf0, acc[it][0]);
      acc[it][1] = wmma_bf16(a, bf1, acc[it][1]);
    }
  }

  const int lane = tid & 31, nn = lane & 15, hi = lane >> 4;
#pragma unroll
  for (int it = 0; it < 4; ++it) {
#pragma unroll
    for (int jt = 0; jt < 2; ++jt) {
      int o = n0 + wn * 32 + jt * 16 + nn;
      float bv = bias[o];
      int h = o >> 6, dd = o & 63;
#pragma unroll
      for (int r = 0; r < 8; ++r) {
        int m = m0 + wm * 64 + it * 16 + r + 8 * hi;
        int b = m >> 10, n = m & 1023;
        float val = acc[it][jt][r] + bv;
        if (MODE == 0) {
          val *= 0.125f;  // 1/sqrt(64) folded into Q
          dst[(((size_t)b * 8 + h) * 1024 + n) * 64 + dd] = to_bf16(val);
        } else if (MODE == 1) {
          dst[(((size_t)b * 8 + h) * 1024 + n) * 64 + dd] = to_bf16(val);
        } else {
          dst[(((size_t)b * 8 + h) * 64 + dd) * 1024 + n] = to_bf16(val);
        }
      }
    }
  }
}

// ---------------- Kernel 3: attention (scores + softmax + P*V) ----------------
// One WG per (b*8+h, 32-query block). Dynamic LDS:
//   S[32][1024] f32 (128KB) + P[32][1024] bf16 (64KB) + Q[32][64] bf16 (4KB)
//   + lrow[32] f32  = 200832 bytes (WGP LDS is 320KB).
__global__ __launch_bounds__(256) void attn_kernel(const bf16* __restrict__ q,
                                                   const bf16* __restrict__ k,
                                                   const bf16* __restrict__ v,
                                                   bf16* __restrict__ ctx) {
  extern __shared__ char smem[];
  float* S    = (float*)smem;                                   // [32][1024] f32
  bf16*  P    = (bf16*)(smem + 131072);                         // [32][1024] bf16
  bf16*  Qs   = (bf16*)(smem + 131072 + 65536);                 // [32][64]
  float* lrow = (float*)(smem + 131072 + 65536 + 4096);         // [32]

  const int tid = threadIdx.x;
  const int bh = blockIdx.x >> 5;            // b*8+h, 64 total
  const int i0 = (blockIdx.x & 31) * 32;     // query block origin
  const bf16* qp = q + ((size_t)bh * 1024 + i0) * 64;
  const bf16* kp = k + (size_t)bh * 1024 * 64;
  const bf16* vp = v + (size_t)bh * 64 * 1024;

  // Stage 32x64 Q tile (pre-scaled)
  copy16_g2l(&qp[tid * 8], &Qs[tid * 8]);
  async_copy_wait();
  __syncthreads();

  const int w = tid >> 5, lane = tid & 31;
  const int it = w & 1, jw = w >> 1, nn = lane & 15, hi = lane >> 4;

  v16bf a0 = load_frag_a(&Qs[(it * 16) * 64], 64);        // d = 0..31
  v16bf a1 = load_frag_a(&Qs[(it * 16) * 64 + 32], 64);   // d = 32..63

  // Phase 1: S = Q K^T (K read from global; L2-resident)
  for (int s = 0; s < 16; ++s) {
    int jt = jw + 4 * s;
    const bf16* kb = kp + (size_t)(jt * 16) * 64;
    if (s + 1 < 16)
      __builtin_prefetch(kp + (size_t)((jt + 4) * 16) * 64, 0, 1);
    v16bf b0 = load_frag_b(kb, 64);
    v16bf b1 = load_frag_b(kb + 32, 64);
    v8f c;
#pragma unroll
    for (int e = 0; e < 8; ++e) c[e] = 0.f;
    c = wmma_bf16(a0, b0, c);
    c = wmma_bf16(a1, b1, c);
#pragma unroll
    for (int r = 0; r < 8; ++r)
      S[(it * 16 + r + 8 * hi) * 1024 + jt * 16 + nn] = c[r];
  }
  __syncthreads();

  // Phase 2: row softmax; write exp() as bf16 into P, keep row sums
  {
    const int row = tid >> 3, sub = tid & 7;    // 8 lanes per row, same wave
    float* Sr = S + row * 1024;
    bf16*  Pr = P + row * 1024;
    float m = -3.0e38f;
    for (int j = sub; j < 1024; j += 8) m = fmaxf(m, Sr[j]);
#pragma unroll
    for (int mk = 4; mk >= 1; mk >>= 1) m = fmaxf(m, __shfl_xor(m, mk, 32));
    float l = 0.f;
    for (int j = sub; j < 1024; j += 8) {
      float e = __expf(Sr[j] - m);
      Pr[j] = to_bf16(e);
      l += e;
    }
#pragma unroll
    for (int mk = 4; mk >= 1; mk >>= 1) l += __shfl_xor(l, mk, 32);
    if (sub == 0) lrow[row] = l;
  }
  __syncthreads();

  // Phase 3: O = P * V ; wave tile rows it*16, cols dt*16 of d
  const int dt = w >> 1;
  v8f o;
#pragma unroll
  for (int e = 0; e < 8; ++e) o[e] = 0.f;
  const bf16* vb0 = vp + (size_t)(dt * 16) * 1024;
  for (int j0 = 0; j0 < 1024; j0 += 32) {
    if (j0 + 32 < 1024)
      __builtin_prefetch(vb0 + j0 + 32, 0, 1);
    v16bf pa = load_frag_a(&P[(it * 16) * 1024 + j0], 1024);
    v16bf vb = load_frag_b(vb0 + j0, 1024);
    o = wmma_bf16(pa, vb, o);
  }
  const int b = bh >> 3, h = bh & 7;
#pragma unroll
  for (int r = 0; r < 8; ++r) {
    int ri = it * 16 + r + 8 * hi;
    float val = o[r] / lrow[ri];
    ctx[((size_t)b * 1024 + i0 + ri) * 512 + h * 64 + dt * 16 + nn] = to_bf16(val);
  }
}

// ---------------- Kernel 4: output projection + bias + residual ----------------
__global__ __launch_bounds__(256) void out_gemm_kernel(
    const bf16* __restrict__ ctx, const bf16* __restrict__ wt,
    const float* __restrict__ bias, const float* __restrict__ xres,
    float* __restrict__ out) {
  __shared__ __align__(16) bf16 As[128 * 32];
  __shared__ __align__(16) bf16 Bs[128 * 32];
  const int tid = threadIdx.x;
  const int m0 = (blockIdx.x >> 2) * 128;
  const int n0 = (blockIdx.x & 3) * 128;
  const int w  = tid >> 5;
  const int wm = w & 1;
  const int wn = w >> 1;
  v8f acc[4][2];
#pragma unroll
  for (int i = 0; i < 4; ++i)
#pragma unroll
    for (int j = 0; j < 2; ++j)
#pragma unroll
      for (int e = 0; e < 8; ++e) acc[i][j][e] = 0.f;

  for (int kk = 0; kk < 512; kk += 32) {
    __syncthreads();
    for (int ch = tid; ch < 512; ch += 256) {
      int r = ch >> 2, c8 = (ch & 3) * 8;
      copy16_g2l(&ctx[(size_t)(m0 + r) * 512 + kk + c8], &As[r * 32 + c8]);
      copy16_g2l(&wt[(size_t)(n0 + r) * 512 + kk + c8], &Bs[r * 32 + c8]);
    }
    async_copy_wait();
    __syncthreads();
    v16bf bf0 = load_frag_b(&Bs[(wn * 32 + 0) * 32], 32);
    v16bf bf1 = load_frag_b(&Bs[(wn * 32 + 16) * 32], 32);
#pragma unroll
    for (int it = 0; it < 4; ++it) {
      v16bf a = load_frag_a(&As[(wm * 64 + it * 16) * 32], 32);
      acc[it][0] = wmma_bf16(a, bf0, acc[it][0]);
      acc[it][1] = wmma_bf16(a, bf1, acc[it][1]);
    }
  }

  const int lane = tid & 31, nn = lane & 15, hi = lane >> 4;
#pragma unroll
  for (int it = 0; it < 4; ++it) {
#pragma unroll
    for (int jt = 0; jt < 2; ++jt) {
      int o = n0 + wn * 32 + jt * 16 + nn;
      float bv = bias[o];
#pragma unroll
      for (int r = 0; r < 8; ++r) {
        int m = m0 + wm * 64 + it * 16 + r + 8 * hi;
        int b = m >> 10, n = m & 1023;
        size_t idx = ((size_t)b * 512 + o) * 1024 + n;   // [B][C][H*W]
        out[idx] = xres[idx] + acc[it][jt][r] + bv;
      }
    }
  }
}

extern "C" void kernel_launch(void* const* d_in, const int* in_sizes, int n_in,
                              void* d_out, int out_size, void* d_ws, size_t ws_size,
                              hipStream_t stream) {
  (void)in_sizes; (void)n_in; (void)out_size; (void)ws_size;
  const float* x   = (const float*)d_in[0];
  const float* gnw = (const float*)d_in[1];
  const float* gnb = (const float*)d_in[2];
  const float* wq  = (const float*)d_in[3];
  const float* bq  = (const float*)d_in[4];
  const float* wk  = (const float*)d_in[5];
  const float* bk  = (const float*)d_in[6];
  const float* wv  = (const float*)d_in[7];
  const float* bv  = (const float*)d_in[8];
  const float* wo  = (const float*)d_in[9];
  const float* bo  = (const float*)d_in[10];

  char* ws = (char*)d_ws;
  bf16* xn  = (bf16*)(ws);
  bf16* qb  = (bf16*)(ws + ((size_t)8  << 20));
  bf16* kb  = (bf16*)(ws + ((size_t)16 << 20));
  bf16* vb  = (bf16*)(ws + ((size_t)24 << 20));
  bf16* ctx = (bf16*)(ws + ((size_t)32 << 20));
  bf16* wqb = (bf16*)(ws + ((size_t)40 << 20));
  bf16* wkb = wqb + 512 * 512;
  bf16* wvb = wkb + 512 * 512;
  bf16* wob = wvb + 512 * 512;

  wcvt_kernel<<<1024, 256, 0, stream>>>(wq, wk, wv, wo, wqb, wkb, wvb, wob);
  gn_kernel<<<256, 256, 0, stream>>>(x, gnw, gnb, xn);
  qkv_gemm_kernel<0><<<256, 256, 0, stream>>>(xn, wqb, bq, qb);
  qkv_gemm_kernel<1><<<256, 256, 0, stream>>>(xn, wkb, bk, kb);
  qkv_gemm_kernel<2><<<256, 256, 0, stream>>>(xn, wvb, bv, vb);
  size_t attn_smem = 131072 + 65536 + 4096 + 128;  // 200832 B
  attn_kernel<<<2048, 256, attn_smem, stream>>>(qb, kb, vb, ctx);
  out_gemm_kernel<<<256, 256, 0, stream>>>(ctx, wob, bo, x, (float*)d_out);
}
